// SageSparseLinearAttention_4440996184230
// MI455X (gfx1250) — compile-verified, matrix operand order checked by reading, not run on hardware
//
#include <hip/hip_runtime.h>
#include <cstddef>
#include <cstdint>

typedef __attribute__((ext_vector_type(16))) _Float16 v16h;
typedef __attribute__((ext_vector_type(8)))  float    v8f;

#define H_    16
#define L_    8192
#define D_    64
#define M_    128   // L / BLKQ
#define N_    256   // L / BLKK
#define T_    25    // top-k key blocks
#define BLKQ_ 64
#define BLKK_ 32

__device__ __forceinline__ v8f wmma_f16f32(v16h a, v16h b, v8f c) {
  // D = A(16x32 f16) * B(32x16 f16) + C(16x16 f32)
  return __builtin_amdgcn_wmma_f32_16x16x32_f16(false, a, false, b, (short)0, c,
                                                false, false);
}

// pack two floats into 2 x f16 in a dword
__device__ __forceinline__ uint32_t pkh(float a, float b) {
  union { _Float16 h[2]; uint32_t u; } x;
  x.h[0] = (_Float16)a;
  x.h[1] = (_Float16)b;
  return x.u;
}

// DPP ROW_XMASK lane-xor within each 16-lane row (VALU, no LDS traffic)
template <int MASK>
__device__ __forceinline__ float dpp_xor(float x) {
  int xi = __float_as_int(x);
  int yi = __builtin_amdgcn_update_dpp(xi, xi, 0x160 | MASK, 0xf, 0xf, false);
  return __int_as_float(yi);
}
// half-wave (16-lane) reductions
__device__ __forceinline__ float hmax16(float x) {
  x = fmaxf(x, dpp_xor<1>(x));
  x = fmaxf(x, dpp_xor<2>(x));
  x = fmaxf(x, dpp_xor<4>(x));
  x = fmaxf(x, dpp_xor<8>(x));
  return x;
}
__device__ __forceinline__ float hsum16(float x) {
  x += dpp_xor<1>(x);
  x += dpp_xor<2>(x);
  x += dpp_xor<4>(x);
  x += dpp_xor<8>(x);
  return x;
}
// uniform broadcast of lane `l` (compile-time) -> v_readlane + scalar operand
__device__ __forceinline__ float bcast(float x, int l) {
  return __int_as_float(__builtin_amdgcn_readlane(__float_as_int(x), l));
}

// Fragment loader: lane indexes the STRIDED dim; elements walk the unit dim.
// Both 8-half runs are 16B-aligned -> ds_load_b128 pairs.
__device__ __forceinline__ v16h frag_lr(const _Float16* p, int stride, int rbase,
                                        int cbase) {
  const int lane = threadIdx.x & 31;
  const int r = rbase + (lane & 15);
  const int koff = (lane >> 4) * 8;
  v16h f;
#pragma unroll
  for (int e = 0; e < 8; ++e) {
    f[e]     = p[r * stride + cbase + koff + e];
    f[e + 8] = p[r * stride + cbase + 16 + koff + e];
  }
  return f;
}

// ---------------- per-head key mean ----------------
__global__ __launch_bounds__(64) void km_kernel(const float* __restrict__ k,
                                                float* __restrict__ km) {
  const int h = blockIdx.x;
  const int part = blockIdx.y;          // 8 L-partitions
  const int d = threadIdx.x;
  float s = 0.f;
  for (int i = 0; i < L_ / 8; ++i) {
    int l = part * (L_ / 8) + i;
    s += k[((size_t)l * H_ + h) * D_ + d];
  }
  atomicAdd(&km[h * D_ + d], s * (1.0f / (float)L_));
}

// ---------------- block pooling ----------------
__global__ __launch_bounds__(64) void pooledq_kernel(const float* __restrict__ q,
                                                     float* __restrict__ pq) {
  const int h = blockIdx.x, m = blockIdx.y, d = threadIdx.x;
  float s = 0.f;
  for (int r = 0; r < BLKQ_; ++r)
    s += q[((size_t)(m * BLKQ_ + r) * H_ + h) * D_ + d];
  pq[((size_t)h * M_ + m) * D_ + d] = s * (1.0f / (float)BLKQ_);
}

__global__ __launch_bounds__(64) void pooledk_kernel(const float* __restrict__ k,
                                                     const float* __restrict__ km,
                                                     float* __restrict__ pk) {
  const int h = blockIdx.x, n = blockIdx.y, d = threadIdx.x;
  float s = 0.f;
  for (int r = 0; r < BLKK_; ++r)
    s += k[((size_t)(n * BLKK_ + r) * H_ + h) * D_ + d];
  pk[((size_t)h * N_ + n) * D_ + d] = s * (1.0f / (float)BLKK_) - km[h * D_ + d];
}

// ---------------- routing scores ----------------
__global__ __launch_bounds__(256) void score_kernel(const float* __restrict__ pq,
                                                    const float* __restrict__ pk,
                                                    float* __restrict__ score) {
  const int h = blockIdx.x, m = blockIdx.y;
  const int n = threadIdx.x;            // 256 key blocks
  __shared__ float qrow[D_];
  if (n < D_) qrow[n] = pq[((size_t)h * M_ + m) * D_ + n];
  __syncthreads();
  const float* kp = pk + ((size_t)h * N_ + n) * D_;
  float s = 0.f;
#pragma unroll 8
  for (int d = 0; d < D_; ++d) s += qrow[d] * kp[d];
  score[((size_t)h * M_ + m) * N_ + n] = s;
}

// ---------------- top-25 per row, sorted ascending ----------------
__global__ __launch_bounds__(32) void topk_kernel(const float* __restrict__ score,
                                                  int* __restrict__ lut) {
  const int row = blockIdx.x;           // h*M_ + m
  const int lane = threadIdx.x;
  __shared__ int list[T_];
  float sc[8];
#pragma unroll
  for (int j = 0; j < 8; ++j) sc[j] = score[(size_t)row * N_ + j * 32 + lane];
  for (int it = 0; it < T_; ++it) {
    float bv = -1e30f;
    int bi = N_;
#pragma unroll
    for (int j = 0; j < 8; ++j)
      if (sc[j] > bv) { bv = sc[j]; bi = j * 32 + lane; }
    for (int d = 1; d < 32; d <<= 1) {
      float ov = __shfl_xor(bv, d);
      int oi = __shfl_xor(bi, d);
      if (ov > bv || (ov == bv && oi < bi)) { bv = ov; bi = oi; }
    }
#pragma unroll
    for (int j = 0; j < 8; ++j)
      if ((j * 32 + lane) == bi) sc[j] = -1e30f;
    if (lane == 0) list[it] = bi;
  }
  __syncthreads();
  if (lane < T_) {
    int vv = list[lane];
    int rank = 0;
#pragma unroll
    for (int j = 0; j < T_; ++j) rank += (list[j] < vv);
    lut[(size_t)row * T_ + rank] = vv;
  }
}

// ---------------- block-sparse flash attention (WMMA f16) ----------------
__global__ __launch_bounds__(128) void sparse_attn_kernel(
    const float* __restrict__ q, const float* __restrict__ k,
    const float* __restrict__ v, const float* __restrict__ km,
    const int* __restrict__ lut, float* __restrict__ out) {
  const int h = blockIdx.x;
  const int m = blockIdx.y;
  const int tid = threadIdx.x;
  const int lane = tid & 31;
  const int w = tid >> 5;               // 4 waves, 16 Q rows each

  __shared__ __align__(16) _Float16 qs[BLKQ_ * 72];     // row-major Q
  __shared__ __align__(16) _Float16 ks[BLKK_ * 72];     // row-major centered K
  __shared__ __align__(16) _Float16 vsT[D_ * 40];       // V transposed [d][key]
  __shared__ __align__(16) _Float16 ps[4][16 * 40];     // per-wave P staging

  // stage Q block (64x64) as packed f16 pairs
  {
    int r = tid >> 1, c0 = (tid & 1) * 32;
    const float* qp = q + ((size_t)(m * BLKQ_ + r) * H_ + h) * D_ + c0;
    uint32_t* dst = (uint32_t*)&qs[r * 72 + c0];
#pragma unroll
    for (int i = 0; i < 16; ++i) dst[i] = pkh(qp[2 * i], qp[2 * i + 1]);
  }
  __syncthreads();

  const v16h aq0 = frag_lr(qs, 72, w * 16, 0);   // Q strip, K = d 0..31
  const v16h aq1 = frag_lr(qs, 72, w * 16, 32);  // Q strip, K = d 32..63

  v8f o0 = {}, o1 = {}, o2 = {}, o3 = {};
  float mrun[8], lrun[8];
#pragma unroll
  for (int j = 0; j < 8; ++j) { mrun[j] = -1e30f; lrun[j] = 0.f; }
  const float scale = 0.125f;           // 1/sqrt(64)
  const int* lrow = lut + ((size_t)h * M_ + m) * T_;

  for (int t = 0; t < T_; ++t) {
    const int n = lrow[t];
    if (t + 1 < T_) {                   // prefetch next gathered block
      int n2 = lrow[t + 1];
      __builtin_prefetch(k + ((size_t)(n2 * BLKK_) * H_ + h) * D_, 0, 1);
      __builtin_prefetch(v + ((size_t)(n2 * BLKK_) * H_ + h) * D_, 0, 1);
    }
    // centered K (row-major): thread -> (key row, 16 cols), packed stores
    {
      int r = tid >> 2, c0 = (tid & 3) * 16;
      const float* kp = k + ((size_t)(n * BLKK_ + r) * H_ + h) * D_ + c0;
      const float* kmp = km + h * D_ + c0;
      uint32_t* dst = (uint32_t*)&ks[r * 72 + c0];
#pragma unroll
      for (int i = 0; i < 8; ++i)
        dst[i] = pkh(kp[2 * i] - kmp[2 * i], kp[2 * i + 1] - kmp[2 * i + 1]);
    }
    // V transposed: thread -> (2 keys, 8 d); pack the two keys per dword
    {
      int kp2 = tid >> 3;               // key pair 0..15
      int dg = (tid & 7) * 8;
      const float* v0p = v + ((size_t)(n * BLKK_ + 2 * kp2) * H_ + h) * D_ + dg;
      const float* v1p = v0p + (size_t)H_ * D_;
#pragma unroll
      for (int i = 0; i < 8; ++i)
        *(uint32_t*)&vsT[(dg + i) * 40 + 2 * kp2] = pkh(v0p[i], v1p[i]);
    }
    __syncthreads();

    // S = Q * K^T  (two 16x16 tiles: key cols 0..15, 16..31)
    v8f s0 = {}, s1 = {};
    s0 = wmma_f16f32(aq0, frag_lr(ks, 72, 0, 0), s0);
    s0 = wmma_f16f32(aq1, frag_lr(ks, 72, 0, 32), s0);
    s1 = wmma_f16f32(aq0, frag_lr(ks, 72, 16, 0), s1);
    s1 = wmma_f16f32(aq1, frag_lr(ks, 72, 16, 32), s1);

    float bm[8];
#pragma unroll
    for (int j = 0; j < 8; ++j) {
      s0[j] *= scale; s1[j] *= scale;
      bm[j] = hmax16(fmaxf(s0[j], s1[j]));   // row max via DPP ROW_XMASK
    }
    float alpha[8], bl[8];
#pragma unroll
    for (int j = 0; j < 8; ++j) {
      float mn = fmaxf(mrun[j], bm[j]);
      alpha[j] = __expf(mrun[j] - mn);
      mrun[j] = mn;
      s0[j] = __expf(s0[j] - mn);
      s1[j] = __expf(s1[j] - mn);
      bl[j] = hsum16(s0[j] + s1[j]);         // row sum via DPP ROW_XMASK
    }
#pragma unroll
    for (int j = 0; j < 8; ++j) {
      lrun[j] = lrun[j] * alpha[j] + bl[j];
      o0[j] *= alpha[j]; o1[j] *= alpha[j]; o2[j] *= alpha[j]; o3[j] *= alpha[j];
    }

    // stage P (C-layout -> row-major f16) through per-wave LDS
    _Float16* pw = &ps[w][0];
    const int prow = 8 * (lane >> 4);
#pragma unroll
    for (int j = 0; j < 8; ++j) {
      pw[(prow + j) * 40 + (lane & 15)]      = (_Float16)s0[j];
      pw[(prow + j) * 40 + 16 + (lane & 15)] = (_Float16)s1[j];
    }
    const v16h ap = frag_lr(pw, 40, 0, 0);   // A = P[16 x 32]

    // O += P * V  (4 tiles over d; B from transposed V -> ds_load_b128)
    o0 = wmma_f16f32(ap, frag_lr(vsT, 40, 0, 0), o0);
    o1 = wmma_f16f32(ap, frag_lr(vsT, 40, 16, 0), o1);
    o2 = wmma_f16f32(ap, frag_lr(vsT, 40, 32, 0), o2);
    o3 = wmma_f16f32(ap, frag_lr(vsT, 40, 48, 0), o3);
    __syncthreads();
  }

  // write O / l  -> out[l][h][d]  (o_s component; overwrites poison)
#pragma unroll
  for (int j = 0; j < 8; ++j) {
    const int row = 16 * w + j + 8 * (lane >> 4);
    const float inv = 1.0f / lrun[j];
    float* op = out + ((size_t)(m * BLKQ_ + row) * H_ + h) * D_ + (lane & 15);
    op[0]  = o0[j] * inv;
    op[16] = o1[j] * inv;
    op[32] = o2[j] * inv;
    op[48] = o3[j] * inv;
  }
}

// ---------------- kvsum = k_fm^T @ V and ksum (WMMA f16) ----------------
__global__ __launch_bounds__(128) void kvsum_kernel(const float* __restrict__ k,
                                                    const float* __restrict__ v,
                                                    float* __restrict__ kvsum,
                                                    float* __restrict__ ksum) {
  const int h = blockIdx.x;
  const int s = blockIdx.y;             // 8 L-splits
  const int tid = threadIdx.x;
  const int lane = tid & 31;
  const int w = tid >> 5;

  __shared__ __align__(16) _Float16 kfT[D_ * 40];   // k_fm transposed [d][l]
  __shared__ __align__(16) _Float16 vfT[D_ * 40];   // V transposed   [e][l]

  const int g = tid >> 3;               // 16 groups -> rows 2g, 2g+1
  const int jj = tid & 7;
  const int dg = jj * 8;                // 8 d-columns per thread
  float ksp[8];
#pragma unroll
  for (int i = 0; i < 8; ++i) ksp[i] = 0.f;
  v8f acc0 = {}, acc1 = {}, acc2 = {}, acc3 = {};

  for (int it = 0; it < (L_ / 8) / 32; ++it) {
    const int l0 = s * (L_ / 8) + it * 32 + 2 * g;
    const float* kp0 = k + ((size_t)l0 * H_ + h) * D_ + dg;
    const float* kp1 = kp0 + (size_t)H_ * D_;
    float a0[8], a1[8];
    float m0 = -1e30f, m1 = -1e30f;
#pragma unroll
    for (int i = 0; i < 8; ++i) {
      a0[i] = kp0[i]; m0 = fmaxf(m0, a0[i]);
      a1[i] = kp1[i]; m1 = fmaxf(m1, a1[i]);
    }
    // reduce across the 8-thread group (DPP xor 1,2,4)
    m0 = fmaxf(m0, dpp_xor<1>(m0)); m0 = fmaxf(m0, dpp_xor<2>(m0));
    m0 = fmaxf(m0, dpp_xor<4>(m0));
    m1 = fmaxf(m1, dpp_xor<1>(m1)); m1 = fmaxf(m1, dpp_xor<2>(m1));
    m1 = fmaxf(m1, dpp_xor<4>(m1));
    float s0 = 0.f, s1 = 0.f;
#pragma unroll
    for (int i = 0; i < 8; ++i) {
      a0[i] = __expf(a0[i] - m0); s0 += a0[i];
      a1[i] = __expf(a1[i] - m1); s1 += a1[i];
    }
    s0 += dpp_xor<1>(s0); s0 += dpp_xor<2>(s0); s0 += dpp_xor<4>(s0);
    s1 += dpp_xor<1>(s1); s1 += dpp_xor<2>(s1); s1 += dpp_xor<4>(s1);
    const float i0 = 1.0f / s0, i1 = 1.0f / s1;
    const float* vp0 = v + ((size_t)l0 * H_ + h) * D_ + dg;
    const float* vp1 = vp0 + (size_t)H_ * D_;
#pragma unroll
    for (int i = 0; i < 8; ++i) {
      float f0 = a0[i] * i0, f1 = a1[i] * i1;
      ksp[i] += f0 + f1;
      *(uint32_t*)&kfT[(dg + i) * 40 + 2 * g] = pkh(f0, f1);
      *(uint32_t*)&vfT[(dg + i) * 40 + 2 * g] = pkh(vp0[i], vp1[i]);
    }
    __syncthreads();
    // A = k_fm^T strip [16 d x 32 l]; B = V [32 l x 16 e] (both row pattern)
    const v16h a = frag_lr(kfT, 40, 16 * w, 0);
    acc0 = wmma_f16f32(a, frag_lr(vfT, 40, 0, 0), acc0);
    acc1 = wmma_f16f32(a, frag_lr(vfT, 40, 16, 0), acc1);
    acc2 = wmma_f16f32(a, frag_lr(vfT, 40, 32, 0), acc2);
    acc3 = wmma_f16f32(a, frag_lr(vfT, 40, 48, 0), acc3);
    __syncthreads();
  }

#pragma unroll
  for (int j = 0; j < 8; ++j) {
    const int d = 16 * w + j + 8 * (lane >> 4);
    float* kvp = kvsum + (size_t)h * D_ * D_ + d * D_ + (lane & 15);
    atomicAdd(kvp + 0,  acc0[j]);
    atomicAdd(kvp + 16, acc1[j]);
    atomicAdd(kvp + 32, acc2[j]);
    atomicAdd(kvp + 48, acc3[j]);
  }
#pragma unroll
  for (int i = 0; i < 8; ++i) atomicAdd(&ksum[h * D_ + dg + i], ksp[i]);
}

// ---------------- linear branch + projection, out += o_l ----------------
__global__ __launch_bounds__(256) void combine_kernel(
    const float* __restrict__ q, const float* __restrict__ kvsum,
    const float* __restrict__ ksum, const float* __restrict__ W,
    const float* __restrict__ b, float* __restrict__ out) {
  const int gid = blockIdx.x * 8 + (threadIdx.x >> 5);  // row over L*H
  const int lane = threadIdx.x & 31;
  const int l = gid >> 4;
  const int h = gid & 15;
  const int e0 = 2 * lane, e1 = e0 + 1;

  const float* qp = q + ((size_t)l * H_ + h) * D_;
  float q0 = qp[e0], q1 = qp[e1];
  float mx = hmax16(fmaxf(q0, q1));
  mx = fmaxf(mx, __shfl_xor(mx, 16));
  q0 = __expf(q0 - mx);
  q1 = __expf(q1 - mx);
  float sm = hsum16(q0 + q1);
  sm += __shfl_xor(sm, 16);
  const float invs = 1.0f / sm;
  q0 *= invs; q1 *= invs;

  const float* ks = ksum + h * D_;
  float dn = hsum16(q0 * ks[e0] + q1 * ks[e1]);
  dn += __shfl_xor(dn, 16);
  dn += 1e-6f;

  const float* kv = kvsum + (size_t)h * D_ * D_;
  float t0 = 0.f, t1 = 0.f;
#pragma unroll
  for (int sl = 0; sl < 32; ++sl) {
    const float qa = bcast(q0, sl);      // v_readlane -> scalar operand
    const float qb = bcast(q1, sl);
    const int d0 = 2 * sl;
    t0 += qa * kv[d0 * D_ + e0] + qb * kv[(d0 + 1) * D_ + e0];
    t1 += qa * kv[d0 * D_ + e1] + qb * kv[(d0 + 1) * D_ + e1];
  }
  const float invd = 1.0f / dn;
  const float ol0 = t0 * invd, ol1 = t1 * invd;

  // projection: r[d] = b[d] + sum_e ol[e] * W[d][e]
  float r0 = b[e0], r1 = b[e1];
#pragma unroll
  for (int sl = 0; sl < 32; ++sl) {
    const float oa = bcast(ol0, sl);
    const float ob = bcast(ol1, sl);
    const int ee = 2 * sl;
    r0 += oa * W[e0 * D_ + ee] + ob * W[e0 * D_ + ee + 1];
    r1 += oa * W[e1 * D_ + ee] + ob * W[e1 * D_ + ee + 1];
  }
  float* op = out + ((size_t)l * H_ + h) * D_;
  op[e0] += r0;
  op[e1] += r1;
}

extern "C" void kernel_launch(void* const* d_in, const int* in_sizes, int n_in,
                              void* d_out, int out_size, void* d_ws,
                              size_t ws_size, hipStream_t stream) {
  (void)in_sizes; (void)n_in; (void)out_size; (void)ws_size;
  const float* q = (const float*)d_in[0];
  const float* k = (const float*)d_in[1];
  const float* v = (const float*)d_in[2];
  const float* W = (const float*)d_in[3];
  const float* b = (const float*)d_in[4];
  float* out = (float*)d_out;

  char* ws = (char*)d_ws;
  const size_t KM_B  = (size_t)H_ * D_ * 4;                 // 4 KB
  const size_t PQ_B  = (size_t)H_ * M_ * D_ * 4;            // 512 KB
  const size_t PK_B  = (size_t)H_ * N_ * D_ * 4;            // 1 MB
  const size_t SC_B  = (size_t)H_ * M_ * N_ * 4;            // 2 MB
  const size_t LUT_B = (size_t)H_ * M_ * T_ * 4;            // 200 KB
  const size_t KV_B  = (size_t)H_ * D_ * D_ * 4;            // 256 KB
  float* km    = (float*)(ws);
  float* pq    = (float*)(ws + KM_B);
  float* pk    = (float*)(ws + KM_B + PQ_B);
  float* score = (float*)(ws + KM_B + PQ_B + PK_B);
  int*   lut   = (int*)  (ws + KM_B + PQ_B + PK_B + SC_B);
  float* kvsum = (float*)(ws + KM_B + PQ_B + PK_B + SC_B + LUT_B);
  float* ksum  = (float*)(ws + KM_B + PQ_B + PK_B + SC_B + LUT_B + KV_B);

  hipMemsetAsync(km, 0, KM_B, stream);
  hipMemsetAsync(kvsum, 0, KV_B, stream);
  hipMemsetAsync(ksum, 0, KM_B, stream);

  km_kernel<<<dim3(H_, 8), 64, 0, stream>>>(k, km);
  pooledq_kernel<<<dim3(H_, M_), 64, 0, stream>>>(q, pq);
  pooledk_kernel<<<dim3(H_, N_), 64, 0, stream>>>(k, km, pk);
  score_kernel<<<dim3(H_, M_), 256, 0, stream>>>(pq, pk, score);
  topk_kernel<<<H_ * M_, 32, 0, stream>>>(score, lut);
  sparse_attn_kernel<<<dim3(H_, M_), 128, 0, stream>>>(q, k, v, km, lut, out);
  kvsum_kernel<<<dim3(H_, 8), 128, 0, stream>>>(k, v, kvsum, ksum);
  combine_kernel<<<(L_ * H_) / 8, 256, 0, stream>>>(q, kvsum, ksum, W, b, out);
}
